// MoENaive_80169859547414
// MI455X (gfx1250) — matrix operand order repaired and checked by reading, compiled, not run on hardware
//
#include <hip/hip_runtime.h>
#include <hip/hip_bf16.h>
#include <math.h>

// Problem constants (match reference)
#define E_ 8
#define T_ 32768
#define D_ 1024

typedef __bf16 bf16_t;
typedef bf16_t v16bf __attribute__((ext_vector_type(16)));
typedef float  v8f   __attribute__((ext_vector_type(8)));
typedef unsigned int u32x4 __attribute__((ext_vector_type(4)));  // trivial 16B chunk

union FragBF { v16bf v; u32x4 q[2]; };

// ---------------------------------------------------------------------------
// Kernel 1: router. One wave32 per token: 8 dot products vs router rows,
// wave reduction, softmax over 8, top-2 -> dense gate[T][8]. Also converts the
// token row to bf16 (fused, saves a pass over 128MB).
// ---------------------------------------------------------------------------
__global__ __launch_bounds__(256) void router_kernel(
    const float* __restrict__ tokens, const float* __restrict__ rw,
    float* __restrict__ gate, bf16_t* __restrict__ xb)
{
  const int lane = threadIdx.x & 31;
  const int t = blockIdx.x * 8 + (threadIdx.x >> 5);
  const float* xrow = tokens + (size_t)t * D_;

  float acc[E_];
#pragma unroll
  for (int e = 0; e < E_; ++e) acc[e] = 0.f;

#pragma unroll 4
  for (int i = 0; i < D_ / 32; ++i) {
    const int c = lane + 32 * i;
    const float xv = xrow[c];
    xb[(size_t)t * D_ + c] = (bf16_t)xv;
#pragma unroll
    for (int e = 0; e < E_; ++e) acc[e] += xv * rw[e * D_ + c];
  }

#pragma unroll
  for (int off = 16; off > 0; off >>= 1) {
#pragma unroll
    for (int e = 0; e < E_; ++e) acc[e] += __shfl_xor(acc[e], off, 32);
  }

  if (lane == 0) {
    float mx = acc[0];
#pragma unroll
    for (int e = 1; e < E_; ++e) mx = fmaxf(mx, acc[e]);
    float p[E_], s = 0.f;
#pragma unroll
    for (int e = 0; e < E_; ++e) { p[e] = expf(acc[e] - mx); s += p[e]; }
    const float inv = 1.f / s;
#pragma unroll
    for (int e = 0; e < E_; ++e) p[e] *= inv;
    // top-2 (ties -> lower index first, same as lax.top_k)
    int i1 = 0; float v1 = p[0];
#pragma unroll
    for (int e = 1; e < E_; ++e) if (p[e] > v1) { v1 = p[e]; i1 = e; }
    int i2 = -1; float v2 = -1.f;
#pragma unroll
    for (int e = 0; e < E_; ++e) if (e != i1 && p[e] > v2) { v2 = p[e]; i2 = e; }
#pragma unroll
    for (int e = 0; e < E_; ++e)
      gate[(size_t)t * E_ + e] = (e == i1) ? v1 : ((e == i2) ? v2 : 0.f);
  }
}

// ---------------------------------------------------------------------------
// Kernel 2: fp32 -> bf16 convert + transpose weights (w[e][k][n] -> wt[e][n][k])
// so the GEMM's B tiles are row-contiguous in K. 64x64 tile through LDS.
// ---------------------------------------------------------------------------
__global__ __launch_bounds__(256) void transpose_bf16_kernel(
    const float* __restrict__ W, bf16_t* __restrict__ Wt)
{
  __shared__ bf16_t tile[64][72];  // +8 pad to stagger banks
  const int e  = blockIdx.z;
  const int kb = blockIdx.x * 64;
  const int nb = blockIdx.y * 64;
  const int r  = threadIdx.x >> 2;        // 0..63
  const int cs = (threadIdx.x & 3) * 16;  // 0/16/32/48

  const float* src = W + ((size_t)e * D_ + kb) * D_ + nb;
#pragma unroll
  for (int j = 0; j < 16; ++j)
    tile[r][cs + j] = (bf16_t)src[(size_t)r * D_ + cs + j];
  __syncthreads();

  bf16_t* dst = Wt + ((size_t)e * D_ + nb) * D_ + kb;  // Wt[e][n][k]
  bf16_t tmp[16];
#pragma unroll
  for (int j = 0; j < 16; ++j) tmp[j] = tile[cs + j][r];
#pragma unroll
  for (int j = 0; j < 16; ++j) dst[(size_t)r * D_ + cs + j] = tmp[j];
}

// ---------------------------------------------------------------------------
// Kernel 3: WMMA GEMM, 128x128 block tile, BK=32, 8 waves of 64x32 each.
// Double-buffered LDS (one barrier per k-step) + L2 prefetch two steps ahead.
// mode 0: H = gelu(A @ W1e)                -> bf16 Hout
// mode 1: out (+)= gate[:,e] * (A @ W2e)   -> f32 Oout (expert 0 overwrites)
// Bt is the pre-transposed [n][k] bf16 weight slab for this expert.
// ---------------------------------------------------------------------------
__global__ __launch_bounds__(256) void moe_gemm_kernel(
    const bf16_t* __restrict__ A,    // [T, D] bf16 row-major
    const bf16_t* __restrict__ Bt,   // [D(n), D(k)] bf16
    const float*  __restrict__ gate, // [T, E] (mode 1)
    bf16_t* __restrict__ Hout,       // mode 0
    float*  __restrict__ Oout,       // mode 1
    int expert, int mode)
{
  __shared__ __align__(16) bf16_t As[2][128 * 32];  // 2 x 8KB
  __shared__ __align__(16) bf16_t Bs[2][128 * 32];  // 2 x 8KB

  const int tid  = threadIdx.x;
  const int lane = tid & 31;
  const int half = lane >> 4;   // lane 0-15 vs 16-31
  const int lrow = lane & 15;
  const int wave = tid >> 5;            // 0..7
  const int wm = (wave & 1) * 64;       // wave M offset
  const int wn = (wave >> 1) * 32;      // wave N offset
  const int m0 = blockIdx.x * 128;
  const int n0 = blockIdx.y * 128;

  v8f acc[4][2];
#pragma unroll
  for (int i = 0; i < 4; ++i)
#pragma unroll
    for (int j = 0; j < 2; ++j) acc[i][j] = {};

  // cooperative tile loads: 256 threads, 2 threads/row, 32B (16 bf16) each
  const int ldr = tid >> 1;
  const int ldc = (tid & 1) * 16;
  const size_t aBase = (size_t)(m0 + ldr) * D_ + ldc;
  const size_t bBase = (size_t)(n0 + ldr) * D_ + ldc;

  // prologue: stage k-tile 0 into buffer 0
  {
    const u32x4* ag = (const u32x4*)(A + aBase);
    const u32x4* bg = (const u32x4*)(Bt + bBase);
    u32x4 a0 = ag[0], a1 = ag[1];
    u32x4 b0 = bg[0], b1 = bg[1];
    u32x4* as = (u32x4*)(As[0] + ldr * 32 + ldc);
    u32x4* bs = (u32x4*)(Bs[0] + ldr * 32 + ldc);
    as[0] = a0; as[1] = a1;
    bs[0] = b0; bs[1] = b1;
  }
  __syncthreads();

  const int NT = D_ / 32;  // 32 k-steps
  for (int kt = 0; kt < NT; ++kt) {
    const int cur = kt & 1;
    const bool more = (kt + 1 < NT);

    // issue global loads for tile kt+1 (consumed after the WMMAs below)
    u32x4 a0, a1, b0, b1;
    if (more) {
      const u32x4* ag = (const u32x4*)(A + aBase + (kt + 1) * 32);
      const u32x4* bg = (const u32x4*)(Bt + bBase + (kt + 1) * 32);
      a0 = ag[0]; a1 = ag[1];
      b0 = bg[0]; b1 = bg[1];
      if (kt + 2 < NT) {  // warm L2 two tiles ahead (global_prefetch_b8)
        __builtin_prefetch(A + aBase + (kt + 2) * 32, 0, 1);
        __builtin_prefetch(Bt + bBase + (kt + 2) * 32, 0, 1);
      }
    }

    // B fragments (32x16 K-major): lane half 0 holds K0..15, half 1 K16..31
    FragBF bf[2];
#pragma unroll
    for (int nt = 0; nt < 2; ++nt) {
      const bf16_t* bp = Bs[cur] + (wn + nt * 16 + lrow) * 32 + half * 16;
      bf[nt].q[0] = ((const u32x4*)bp)[0];
      bf[nt].q[1] = ((const u32x4*)bp)[1];
    }
#pragma unroll
    for (int mt = 0; mt < 4; ++mt) {
      // A fragment (16x32): half 0 -> K0..15, half 1 -> K16..31
      FragBF af;
      const bf16_t* ap = As[cur] + (wm + mt * 16 + lrow) * 32 + half * 16;
      af.q[0] = ((const u32x4*)ap)[0];
      af.q[1] = ((const u32x4*)ap)[1];
#pragma unroll
      for (int nt = 0; nt < 2; ++nt) {
        acc[mt][nt] = __builtin_amdgcn_wmma_f32_16x16x32_bf16(
            false, af.v, false, bf[nt].v, (short)0, acc[mt][nt], false, false);
      }
    }

    // stage tile kt+1 into the idle buffer; single barrier covers RAW + WAR
    if (more) {
      u32x4* as = (u32x4*)(As[cur ^ 1] + ldr * 32 + ldc);
      u32x4* bs = (u32x4*)(Bs[cur ^ 1] + ldr * 32 + ldc);
      as[0] = a0; as[1] = a1;
      bs[0] = b0; bs[1] = b1;
    }
    __syncthreads();
  }

  // Epilogue. C/D layout: VGPR r, lane -> (M = r + 8*half, N = lrow)
  if (mode == 0) {
#pragma unroll
    for (int mt = 0; mt < 4; ++mt) {
#pragma unroll
      for (int r = 0; r < 8; ++r) {
        const int m = m0 + wm + mt * 16 + half * 8 + r;
#pragma unroll
        for (int nt = 0; nt < 2; ++nt) {
          const int n = n0 + wn + nt * 16 + lrow;
          const float v = acc[mt][nt][r];
          // exact GELU: 0.5*x*(1+erf(x/sqrt(2)))
          const float g = 0.5f * v * (1.f + erff(v * 0.70710678118654752f));
          Hout[(size_t)m * D_ + n] = (bf16_t)g;
        }
      }
    }
  } else {
#pragma unroll
    for (int mt = 0; mt < 4; ++mt) {
#pragma unroll
      for (int r = 0; r < 8; ++r) {
        const int m = m0 + wm + mt * 16 + half * 8 + r;
        const float gv = gate[(size_t)m * E_ + expert];  // hoisted per row
#pragma unroll
        for (int nt = 0; nt < 2; ++nt) {
          const int n = n0 + wn + nt * 16 + lrow;
          const size_t idx = (size_t)m * D_ + n;
          const float val = gv * acc[mt][nt][r];
          if (expert == 0) Oout[idx] = val;   // overwrite poisoned d_out
          else             Oout[idx] += val;  // serialized launches => ok
        }
      }
    }
  }
}

// ---------------------------------------------------------------------------
extern "C" void kernel_launch(void* const* d_in, const int* in_sizes, int n_in,
                              void* d_out, int out_size, void* d_ws, size_t ws_size,
                              hipStream_t stream) {
  (void)in_sizes; (void)n_in; (void)out_size; (void)ws_size;
  const float* tokens = (const float*)d_in[0];
  const float* rw     = (const float*)d_in[1];
  const float* w1     = (const float*)d_in[2];
  const float* w2     = (const float*)d_in[3];
  float* out = (float*)d_out;

  // Workspace carve-up (~161 MB total)
  char* ws = (char*)d_ws;
  bf16_t* xb   = (bf16_t*)ws;                                   // 64 MB
  size_t  off  = (size_t)T_ * D_ * sizeof(bf16_t);
  bf16_t* w1t  = (bf16_t*)(ws + off);                           // 16 MB
  off += (size_t)E_ * D_ * D_ * sizeof(bf16_t);
  bf16_t* w2t  = (bf16_t*)(ws + off);                           // 16 MB
  off += (size_t)E_ * D_ * D_ * sizeof(bf16_t);
  float*  gate = (float*)(ws + off);                            // 1 MB
  off += (size_t)T_ * E_ * sizeof(float);
  bf16_t* hb   = (bf16_t*)(ws + off);                           // 64 MB

  router_kernel<<<T_ / 8, 256, 0, stream>>>(tokens, rw, gate, xb);

  dim3 tg(D_ / 64, D_ / 64, E_);
  transpose_bf16_kernel<<<tg, 256, 0, stream>>>(w1, w1t);
  transpose_bf16_kernel<<<tg, 256, 0, stream>>>(w2, w2t);

  dim3 gg(T_ / 128, D_ / 128);
  for (int e = 0; e < E_; ++e) {
    moe_gemm_kernel<<<gg, 256, 0, stream>>>(
        xb, w1t + (size_t)e * D_ * D_, nullptr, hb, nullptr, e, 0);
    moe_gemm_kernel<<<gg, 256, 0, stream>>>(
        hb, w2t + (size_t)e * D_ * D_, gate, nullptr, out, e, 1);
  }
}
